// SwitchTransformer_6562710028477
// MI455X (gfx1250) — matrix operand order repaired and checked by reading, not compiled, not run on hardware
//
#include <hip/hip_runtime.h>

typedef unsigned short u16;
typedef unsigned int   u32;
typedef __attribute__((ext_vector_type(16))) __bf16 v16bf;
typedef __attribute__((ext_vector_type(2)))  __bf16 v2bf;
typedef __attribute__((ext_vector_type(2)))  float  v2f;
typedef __attribute__((ext_vector_type(8)))  float  v8f;

#define N_TOK 131072
#define CAPS  8192
#define NB_BLK 512   // N_TOK / 256

// ---- workspace layout (bytes) ----
#define W1T_OFF  0u          // [16][128][512] bf16  (2 MB)  W1^T per expert
#define W2T_OFF  2097152u    // [16][512][128] bf16  (2 MB)  W2^T per expert
#define G1H_OFF  4194304u    // [64][512] bf16 hi    (64 KB) Wg1^T split-high
#define G1L_OFF  4259840u    // [64][512] bf16 lo    (64 KB) Wg1^T split-low
#define TOP_OFF  4325376u    // [N] int  top expert
#define KEEP_OFF 4849664u    // [N] int  keep flag
#define SLOT_OFF 5373952u    // [16][CAP] int token id per slot
#define SUM_OFF  5898240u    // [N] float sum(exp) per kept token
#define CNTS_OFF 6422528u    // [512][16] per-block counts
#define OFFS_OFF 6455296u    // [512][16] exclusive block offsets
#define CNTG_OFF 6488064u    // [16] capped totals

union BF {
  v16bf v;
  uint4 q[2];
  u32   u[8];
};

// f32 pair -> packed bf16 (RNE); backend selects native packed cvt if available
__device__ __forceinline__ u32 pk2(float a, float b) {
  v2f f; f.x = a; f.y = b;
  union { v2bf v; u32 u; } c;
  c.v = __builtin_convertvector(f, v2bf);
  return c.u;
}
__device__ __forceinline__ u16 cvt1(float a) {
  union { __bf16 b; u16 u; } c;
  c.b = (__bf16)a;
  return c.u;
}
__device__ __forceinline__ float shflx(float v, int m) { // xor-shuffle within wave32
  int li = (int)(threadIdx.x & 31u);
  return __uint_as_float((u32)__builtin_amdgcn_ds_bpermute(((li ^ m) << 2),
                                                           (int)__float_as_uint(v)));
}
__device__ __forceinline__ int shflb(int v, int src) {   // broadcast from lane 'src'
  return __builtin_amdgcn_ds_bpermute(src << 2, v);
}
// B fragment: bf16 p[0..7] and p[16..23] (column n of B, K-contiguous transposed storage)
__device__ __forceinline__ v16bf ldB16(const u16* p) {
  BF r;
  r.q[0] = *(const uint4*)p;
  r.q[1] = *(const uint4*)(p + 16);
  return r.v;
}
// A fragment from bf16 row: v0-3 = K[kb..kb+7], v4-7 = K[kb+16..kb+23]
__device__ __forceinline__ v16bf ldA16(const u16* row, int kb) {
  BF r;
  r.q[0] = *(const uint4*)(row + kb);
  r.q[1] = *(const uint4*)(row + kb + 16);
  return r.v;
}

// ---------------- K0: weight conversion / transposition ----------------
__global__ void k0_prep(const float* __restrict__ Wg1, const float* __restrict__ W1,
                        const float* __restrict__ W2,
                        u16* __restrict__ g1h, u16* __restrict__ g1l,
                        u16* __restrict__ w1t, u16* __restrict__ w2t) {
  const int T1 = 16 * 128 * 512;
  const int T2 = T1 + 16 * 512 * 128;
  const int T3 = T2 + 64 * 512;
  for (int idx = blockIdx.x * 256 + threadIdx.x; idx < T3; idx += gridDim.x * 256) {
    if (idx < T1) {                       // W1[e,d,h] -> w1t[e][h][d]
      int e = idx >> 16, rem = idx & 65535, h = rem >> 9, d = rem & 511;
      w1t[idx] = cvt1(W1[e * 65536 + d * 128 + h]);
    } else if (idx < T2) {                // W2[e,h,o] -> w2t[e][o][h]
      int i = idx - T1;
      int e = i >> 16, rem = i & 65535, o = rem >> 7, h = rem & 127;
      w2t[i] = cvt1(W2[e * 65536 + h * 512 + o]);
    } else {                              // Wg1[d,j] -> split hi/lo [j][d]
      int i = idx - T2;
      int j = i >> 9, d = i & 511;
      float f = Wg1[d * 64 + j];
      u16 hb = cvt1(f);
      g1h[i] = hb;
      g1l[i] = cvt1(f - __uint_as_float(((u32)hb) << 16));
    }
  }
}

// ---------------- K1: gating (split-bf16 WMMA, f32-accurate) + argmax ----------------
__global__ __launch_bounds__(256) void k1_gate(
    const float* __restrict__ x, const u16* __restrict__ g1h, const u16* __restrict__ g1l,
    const float* __restrict__ bg1, const float* __restrict__ Wg2,
    const float* __restrict__ bg2, int* __restrict__ topE) {
  __shared__ u16   whs[64 * 520];
  __shared__ u16   wls[64 * 520];
  __shared__ float wg2s[1024];
  __shared__ float gs[8][16 * 65];

  int tid = threadIdx.x, lane = tid & 31, wv = tid >> 5;
  int hl = lane >> 4, col = lane & 15;

  {
    const u32* sh = (const u32*)g1h;
    const u32* sl = (const u32*)g1l;
    u32* dh = (u32*)whs; u32* dl = (u32*)wls;
    for (int i = tid; i < 64 * 256; i += 256) {
      int r = i >> 8, c = i & 255;
      dh[r * 260 + c] = sh[i];
      dl[r * 260 + c] = sl[i];
    }
    for (int i = tid; i < 1024; i += 256) wg2s[i] = Wg2[i];
  }
  __syncthreads();

  for (int j = 0; j < 2; ++j) {
    int g = blockIdx.x * 16 + wv + 8 * j;          // 512 blocks * 16 = 8192 groups
    const float* xr = x + (size_t)(g * 16 + col) * 512;
    v8f acc[4] = {};
    for (int kc = 0; kc < 16; ++kc) {
      int kb = kc * 32 + 8 * hl;
      float4 f0 = *(const float4*)(xr + kb);
      float4 f1 = *(const float4*)(xr + kb + 4);
      float4 f2 = *(const float4*)(xr + kb + 16);
      float4 f3 = *(const float4*)(xr + kb + 20);
      float xv[16] = {f0.x, f0.y, f0.z, f0.w, f1.x, f1.y, f1.z, f1.w,
                      f2.x, f2.y, f2.z, f2.w, f3.x, f3.y, f3.z, f3.w};
      BF Ah, Al;
      for (int i = 0; i < 8; ++i) {
        u32 hp = pk2(xv[2 * i], xv[2 * i + 1]);
        float h0 = __uint_as_float(hp << 16);
        float h1 = __uint_as_float(hp & 0xffff0000u);
        Ah.u[i] = hp;
        Al.u[i] = pk2(xv[2 * i] - h0, xv[2 * i + 1] - h1);
      }
      const u16* bh = whs + kc * 32 + 8 * hl;
      const u16* bl = wls + kc * 32 + 8 * hl;
      for (int nt = 0; nt < 4; ++nt) {
        v16bf Bh = ldB16(bh + (nt * 16 + col) * 520);
        v16bf Bl = ldB16(bl + (nt * 16 + col) * 520);
        acc[nt] = __builtin_amdgcn_wmma_f32_16x16x32_bf16(false, Ah.v, false, Bh, (short)0, acc[nt], false, false);
        acc[nt] = __builtin_amdgcn_wmma_f32_16x16x32_bf16(false, Ah.v, false, Bl, (short)0, acc[nt], false, false);
        acc[nt] = __builtin_amdgcn_wmma_f32_16x16x32_bf16(false, Al.v, false, Bh, (short)0, acc[nt], false, false);
      }
    }
    float* gw = gs[wv];
    for (int nt = 0; nt < 4; ++nt) {
      float bb = bg1[nt * 16 + col];
      for (int v = 0; v < 8; ++v) {
        float t = acc[nt][v] + bb;
        gw[(v + 8 * hl) * 65 + nt * 16 + col] = t > 0.f ? t : 0.f;
      }
    }
    asm volatile("s_wait_dscnt 0" ::: "memory");
    if (lane < 16) {                       // second gating layer (64x16) + argmax
      float lg[16];
      for (int q = 0; q < 16; ++q) lg[q] = bg2[q];
      for (int k = 0; k < 64; ++k) {
        float gv = gw[lane * 65 + k];
        for (int q = 0; q < 16; ++q) lg[q] += gv * wg2s[k * 16 + q];
      }
      int best = 0; float bv = lg[0];
      for (int q = 1; q < 16; ++q) if (lg[q] > bv) { bv = lg[q]; best = q; }
      topE[g * 16 + lane] = best;
    }
  }
}

// ---------------- K2/K3/K4: stable routing ----------------
__global__ void k2_count(const int* __restrict__ topE, int* __restrict__ counts) {
  __shared__ int hc[16];
  int tid = threadIdx.x;
  if (tid < 16) hc[tid] = 0;
  __syncthreads();
  int e = topE[blockIdx.x * 256 + tid];
  atomicAdd(&hc[e], 1);
  __syncthreads();
  if (tid < 16) counts[blockIdx.x * 16 + tid] = hc[tid];
}

__global__ void k3_scan(const int* __restrict__ counts, int* __restrict__ offs,
                        int* __restrict__ cntg) {
  int t = threadIdx.x;
  if (t < 16) {
    int run = 0;
    for (int b = 0; b < NB_BLK; ++b) { offs[b * 16 + t] = run; run += counts[b * 16 + t]; }
    cntg[t] = run < CAPS ? run : CAPS;
  }
}

__global__ __launch_bounds__(256) void k4_rank(const int* __restrict__ topE,
                                               const int* __restrict__ offs,
                                               int* __restrict__ keep,
                                               int* __restrict__ slots) {
  __shared__ int wcnt[8][16];
  __shared__ int wbase[8][16];
  int tid = threadIdx.x, lane = tid & 31, wv = tid >> 5;
  int n = blockIdx.x * 256 + tid;
  int e = topE[n];
  u32 mymask = 0;
  for (int j = 0; j < 16; ++j) {
    u32 mj = __builtin_amdgcn_ballot_w32(e == j);
    if (lane == 0) wcnt[wv][j] = __popc(mj);
    if (e == j) mymask = mj;
  }
  int rankw = __popc(mymask & ((1u << lane) - 1u));
  __syncthreads();
  if (tid < 16) {
    int r = 0;
    for (int w = 0; w < 8; ++w) { wbase[w][tid] = r; r += wcnt[w][tid]; }
  }
  __syncthreads();
  int pos = offs[blockIdx.x * 16 + e] + wbase[wv][e] + rankw;
  int kp = pos < CAPS ? 1 : 0;
  keep[n] = kp;
  if (kp) slots[e * CAPS + pos] = n;
}

// ---------------- K5: expert MLP (bf16 WMMA), streaming softmax ----------------
__global__ __launch_bounds__(256) void k5_mlp(
    const float* __restrict__ x, const float* __restrict__ b1g, const float* __restrict__ b2g,
    const u16* __restrict__ w1t, const u16* __restrict__ w2t,
    const int* __restrict__ slots, const int* __restrict__ cntg,
    float* __restrict__ out, float* __restrict__ sumexp) {
  __shared__ u16   w1s[128 * 520];   // 133120 B  W1^T [h][d], padded stride
  __shared__ u16   w2s[512 * 136];   // 139264 B  W2^T [o][h], padded stride
  __shared__ u16   hs[8][16 * 136];  //  34816 B  per-wave h tile
  __shared__ float rss[8][16];

  int e = blockIdx.x & 15, chunk = blockIdx.x >> 4;   // 512 blocks = 16 experts x 32 chunks
  int tid = threadIdx.x, lane = tid & 31, wv = tid >> 5;
  int hl = lane >> 4, col = lane & 15;

  {
    const u32* s1 = (const u32*)(w1t + e * (128 * 512));
    u32* d1 = (u32*)w1s;
    for (int i = tid; i < 128 * 256; i += 256) { int r = i >> 8, c = i & 255; d1[r * 260 + c] = s1[i]; }
    const u32* s2 = (const u32*)(w2t + e * (512 * 128));
    u32* d2 = (u32*)w2s;
    for (int i = tid; i < 512 * 64; i += 256) { int r = i >> 6, c = i & 63; d2[r * 68 + c] = s2[i]; }
  }
  __syncthreads();

  int cnt = cntg[e];
  int gmax = (cnt + 15) >> 4;

  for (int j = 0; j < 2; ++j) {
    int g = chunk * 16 + wv + 8 * j;      // wave-uniform
    if (g >= gmax) continue;
    int prow = g * 16 + col;
    int tok = 0;
    if (prow < cnt) tok = slots[e * CAPS + prow];
    const float* xr = x + (size_t)tok * 512;
    int tokm[8];
    for (int v = 0; v < 8; ++v) tokm[v] = shflb(tok, v + 8 * hl);

    // GEMM1: [16,512] x [512,128]
    v8f acc1[8] = {};
    for (int kc = 0; kc < 16; ++kc) {
      int kb = kc * 32 + 8 * hl;
      float4 f0 = *(const float4*)(xr + kb);
      float4 f1 = *(const float4*)(xr + kb + 4);
      float4 f2 = *(const float4*)(xr + kb + 16);
      float4 f3 = *(const float4*)(xr + kb + 20);
      BF A;
      A.u[0] = pk2(f0.x, f0.y); A.u[1] = pk2(f0.z, f0.w);
      A.u[2] = pk2(f1.x, f1.y); A.u[3] = pk2(f1.z, f1.w);
      A.u[4] = pk2(f2.x, f2.y); A.u[5] = pk2(f2.z, f2.w);
      A.u[6] = pk2(f3.x, f3.y); A.u[7] = pk2(f3.z, f3.w);
      const u16* bb = w1s + kc * 32 + 8 * hl;
      for (int nt = 0; nt < 8; ++nt) {
        v16bf B = ldB16(bb + (nt * 16 + col) * 520);
        acc1[nt] = __builtin_amdgcn_wmma_f32_16x16x32_bf16(false, A.v, false, B, (short)0, acc1[nt], false, false);
      }
    }
    // bias + relu -> bf16 h tile in LDS (C-layout -> A-layout transpose)
    u16* hrow = hs[wv];
    for (int nt = 0; nt < 8; ++nt) {
      float bb = b1g[e * 128 + nt * 16 + col];
      for (int v = 0; v < 8; ++v) {
        float t = acc1[nt][v] + bb;
        t = t > 0.f ? t : 0.f;
        hrow[(v + 8 * hl) * 136 + nt * 16 + col] = cvt1(t);
      }
    }
    asm volatile("s_wait_dscnt 0" ::: "memory");

    v16bf a2[4];
    for (int kc = 0; kc < 4; ++kc) a2[kc] = ldA16(hrow + col * 136, kc * 32 + 8 * hl);

    // GEMM2: [16,128] x [128,512], streamed exp + row sums
    float rsum[8] = {0, 0, 0, 0, 0, 0, 0, 0};
    for (int nb = 0; nb < 32; ++nb) {
      v8f acc = {};
      for (int kc = 0; kc < 4; ++kc) {
        v16bf B = ldB16(w2s + (nb * 16 + col) * 136 + kc * 32 + 8 * hl);
        acc = __builtin_amdgcn_wmma_f32_16x16x32_bf16(false, a2[kc], false, B, (short)0, acc, false, false);
      }
      float bb = b2g[e * 512 + nb * 16 + col];
      for (int v = 0; v < 8; ++v) {
        float t = __expf(acc[v] + bb);
        rsum[v] += t;
        if (g * 16 + v + 8 * hl < cnt)
          out[(size_t)tokm[v] * 512 + nb * 16 + col] = t;   // unnormalized
      }
    }
    for (int v = 0; v < 8; ++v) {
      float s = rsum[v];
      s += shflx(s, 1); s += shflx(s, 2); s += shflx(s, 4); s += shflx(s, 8);
      rsum[v] = s;
    }
    if (col == 0)
      for (int v = 0; v < 8; ++v) rss[wv][v + 8 * hl] = rsum[v];
    asm volatile("s_wait_dscnt 0" ::: "memory");
    if (lane < 16 && prow < cnt) sumexp[tok] = rss[wv][lane];
  }
}

// ---------------- K6: normalize kept rows, zero dropped rows ----------------
__global__ void k6_norm(float* __restrict__ out, const int* __restrict__ keep,
                        const float* __restrict__ sumexp) {
  int idx = blockIdx.x * 256 + threadIdx.x;   // N*128 float4s
  int n = idx >> 7, c = idx & 127;
  float4* p = (float4*)out + (size_t)n * 128 + c;
  if (keep[n]) {
    float s = 1.0f / sumexp[n];
    float4 v = *p;
    v.x *= s; v.y *= s; v.z *= s; v.w *= s;
    *p = v;
  } else {
    float4 z; z.x = z.y = z.z = z.w = 0.f;
    *p = z;
  }
}

extern "C" void kernel_launch(void* const* d_in, const int* in_sizes, int n_in,
                              void* d_out, int out_size, void* d_ws, size_t ws_size,
                              hipStream_t stream) {
  const float* x   = (const float*)d_in[0];
  const float* Wg1 = (const float*)d_in[1];
  const float* bg1 = (const float*)d_in[2];
  const float* Wg2 = (const float*)d_in[3];
  const float* bg2 = (const float*)d_in[4];
  const float* W1  = (const float*)d_in[5];
  const float* b1  = (const float*)d_in[6];
  const float* W2  = (const float*)d_in[7];
  const float* b2  = (const float*)d_in[8];
  float* out = (float*)d_out;
  char*  ws  = (char*)d_ws;

  u16* w1t  = (u16*)(ws + W1T_OFF);
  u16* w2t  = (u16*)(ws + W2T_OFF);
  u16* g1h  = (u16*)(ws + G1H_OFF);
  u16* g1l  = (u16*)(ws + G1L_OFF);
  int* topE = (int*)(ws + TOP_OFF);
  int* keep = (int*)(ws + KEEP_OFF);
  int* slots= (int*)(ws + SLOT_OFF);
  float* sme= (float*)(ws + SUM_OFF);
  int* cnts = (int*)(ws + CNTS_OFF);
  int* offs = (int*)(ws + OFFS_OFF);
  int* cntg = (int*)(ws + CNTG_OFF);

  k0_prep<<<4096, 256, 0, stream>>>(Wg1, W1, W2, g1h, g1l, w1t, w2t);
  k1_gate<<<512, 256, 0, stream>>>(x, g1h, g1l, bg1, Wg2, bg2, topE);
  k2_count<<<NB_BLK, 256, 0, stream>>>(topE, cnts);
  k3_scan<<<1, 32, 0, stream>>>(cnts, offs, cntg);
  k4_rank<<<NB_BLK, 256, 0, stream>>>(topE, offs, keep, slots);
  k5_mlp<<<512, 256, 0, stream>>>(x, b1, b2, w1t, w2t, slots, cntg, out, sme);
  k6_norm<<<65536, 256, 0, stream>>>(out, keep, sme);
}
